// SuperpixelConv_45827301048853
// MI455X (gfx1250) — matrix operand order repaired and checked by reading, not compiled, not run on hardware
//
#include <hip/hip_runtime.h>
#include <math.h>

#define T_   2
#define C_   64
#define H_   160
#define W_   160
#define HW   25600          // H_*W_
#define K2   49             // 7*7
#define NSP  196
#define FAN  3136           // C_*K2
#define NKC  98             // FAN/32 k-chunks for 16x16x32 WMMA

typedef __attribute__((ext_vector_type(16))) _Float16 v16h;
typedef __attribute__((ext_vector_type(8)))  float    v8f;

__device__ __forceinline__ int refl(int i, int n) {
    i = (i < 0) ? -i : i;
    return (i >= n) ? (2 * n - 2 - i) : i;
}

// ---------------- Kernel 1: per-superpixel sums & counts ----------------
__global__ void k_segsum(const float* __restrict__ x, const int* __restrict__ spix,
                         float* __restrict__ sums, float* __restrict__ cnts) {
    int pid = blockIdx.x * blockDim.x + threadIdx.x;
    if (pid >= T_ * HW) return;
    int t = pid / HW, r = pid % HW;
    int s = spix[pid];
    atomicAdd(&cnts[t * NSP + s], 1.0f);
    const float* xb = x + (size_t)t * C_ * HW + r;
    float* sb = sums + ((size_t)t * NSP + s) * C_;
    for (int c = 0; c < C_; ++c) atomicAdd(&sb[c], xb[(size_t)c * HW]);
}

// ---------------- Kernel 2: down = sums/max(cnt,1); sq = |down|^2 -------
__global__ void k_down(float* __restrict__ sums, const float* __restrict__ cnts,
                       float* __restrict__ sq) {
    int i = blockIdx.x * blockDim.x + threadIdx.x;
    if (i >= T_ * NSP) return;
    float inv = 1.0f / fmaxf(cnts[i], 1.0f);
    float acc = 0.f;
    float* d = sums + (size_t)i * C_;
    for (int c = 0; c < C_; ++c) { float v = d[c] * inv; d[c] = v; acc += v * v; }
    sq[i] = acc;
}

// ---------------- Kernel 3: pairwise squared distances ------------------
__global__ void k_pwd(const float* __restrict__ down, const float* __restrict__ sq,
                      float* __restrict__ pwd) {
    int idx = blockIdx.x * blockDim.x + threadIdx.x;
    if (idx >= T_ * NSP * NSP) return;
    int t = idx / (NSP * NSP);
    int r = idx % (NSP * NSP);
    int i = r / NSP, j = r % NSP;
    const float* di = down + ((size_t)t * NSP + i) * C_;
    const float* dj = down + ((size_t)t * NSP + j) * C_;
    float dot = 0.f;
    for (int c = 0; c < C_; ++c) dot += di[c] * dj[c];
    pwd[idx] = fmaxf(sq[t * NSP + i] + sq[t * NSP + j] - 2.0f * dot, 0.0f);
}

// ---------------- Kernel 4: per-pixel scale = 10*softplus(...) ----------
__global__ void k_scale(const float* __restrict__ x, const float* __restrict__ wsc,
                        const float* __restrict__ bsc, float* __restrict__ scl) {
    int pid = blockIdx.x * blockDim.x + threadIdx.x;
    if (pid >= T_ * HW) return;
    int t = pid / HW, r = pid % HW;
    const float* xb = x + (size_t)t * C_ * HW + r;
    float ss = 0.f, dt = 0.f;
    for (int c = 0; c < C_; ++c) {
        float v = xb[(size_t)c * HW];
        ss += v * v;
        dt += wsc[c] * v;
    }
    float v = dt / (sqrtf(ss) + 1e-10f) + bsc[0];
    float sp = (v > 20.f) ? v : logf(1.f + expf(v));
    scl[pid] = 10.f * sp;
}

// ---------------- Kernel 5: pack W_lin into CDNA5 A-fragment layout -----
// Layout: Ap[ot][kc][lane][16 halves], per ISA 16-bit A 16x32 layout:
//   lane<16 : M=lane,    halves = K {0..7, 16..23}
//   lane>=16: M=lane-16, halves = K {8..15, 24..31}
__global__ void k_pack(const float* __restrict__ Wl, _Float16* __restrict__ Ap) {
    int idx = blockIdx.x * blockDim.x + threadIdx.x;
    if (idx >= 4 * NKC * 32 * 16) return;
    int i    = idx & 15;
    int lane = (idx >> 4) & 31;
    int kc   = (idx >> 9) % NKC;
    int ot   = idx / (NKC * 512);
    int o    = ot * 16 + (lane & 15);
    int kin  = ((lane < 16) ? 0 : 8) + ((i < 8) ? i : i + 8);
    int k    = kc * 32 + kin;
    Ap[idx]  = (_Float16)Wl[o * FAN + k];
}

// ---------------- Kernel 6: fused reweighted conv via WMMA --------------
// One wave per 16-pixel strip (row h, cols w0..w0+15). Computes all 64 outputs:
// out(64 x 16) = W(64 x 3136) * B(3136 x 16), B[ck,n] = x[c, nbr_tap(n)]*rw[tap,n]
__global__ void __launch_bounds__(32) k_conv(
    const float* __restrict__ x, const int* __restrict__ spix,
    const float* __restrict__ pwd, const float* __restrict__ scl,
    const _Float16* __restrict__ Ap, const float* __restrict__ blin,
    float* __restrict__ out) {
    __shared__ float rwS[16 * K2];
    __shared__ int   offS[16 * K2];

    int lane = threadIdx.x;
    int w0 = blockIdx.x * 16;
    int h  = blockIdx.y;
    int t  = blockIdx.z;
    const int*   sp = spix + t * HW;
    const float* pw = pwd + (size_t)t * NSP * NSP;

    // Pass 1: unnormalized tap weights e = exp(-scale*sim) + gather offsets
    for (int idx = lane; idx < 16 * K2; idx += 32) {
        int n = idx / K2, tap = idx - n * K2;
        int w  = w0 + n;
        int hh = refl(h + tap / 7 - 3, H_);
        int ww = refl(w + tap % 7 - 3, W_);
        int s  = sp[h * W_ + w];
        int ns = sp[hh * W_ + ww];
        float sim = pw[s * NSP + ns];
        float sc  = scl[t * HW + h * W_ + w];
        rwS[idx]  = expf(-sc * sim);
        offS[idx] = hh * W_ + ww;
    }
    __syncthreads();
    // Pass 2: max-normalize per pixel (lanes 0..15 each own a pixel)
    if (lane < 16) {
        float m = 0.f;
        for (int tap = 0; tap < K2; ++tap) m = fmaxf(m, rwS[lane * K2 + tap]);
        float inv = 1.f / (1e-5f + m);
        for (int tap = 0; tap < K2; ++tap) rwS[lane * K2 + tap] *= inv;
    }
    __syncthreads();

    int n    = lane & 15;
    int rsel = (lane < 16) ? 0 : 8;            // B-frag K group per half-wave
    const float* xb  = x + (size_t)t * C_ * HW;
    const float* rwp = rwS + n * K2;
    const int*   ofp = offS + n * K2;
    const v16h*  Af  = (const v16h*)Ap;        // v16h strides of 16 halves

    v8f acc[4] = {v8f{}, v8f{}, v8f{}, v8f{}};

    for (int kc = 0; kc < NKC; ++kc) {
        // Build B fragment: lane holds K rows (kc*32 + rsel + j + (j&8)), column n
        union { _Float16 hv[16]; v16h v; } B;
        int rowbase = kc * 32 + rsel;
#pragma unroll
        for (int j = 0; j < 16; ++j) {
            int row = rowbase + j + (j & 8);
            int c   = row / K2;
            int tap = row - c * K2;
            B.hv[j] = (_Float16)(xb[(size_t)c * HW + ofp[tap]] * rwp[tap]);
        }
        int abase = kc * 32 + lane;
#pragma unroll
        for (int ot = 0; ot < 4; ++ot) {
            v16h a = Af[ot * NKC * 32 + abase];
            acc[ot] = __builtin_amdgcn_wmma_f32_16x16x32_f16(
                false, a, false, B.v, (short)0, acc[ot], false, false);
        }
    }

    // C/D layout: VGPR v -> M = v (lanes 0-15) or v+8 (lanes 16-31), N = lane%16
    int Mb = (lane < 16) ? 0 : 8;
    float* ob = out + (size_t)t * C_ * HW + h * W_ + (w0 + n);
#pragma unroll
    for (int ot = 0; ot < 4; ++ot) {
#pragma unroll
        for (int v = 0; v < 8; ++v) {
            int o = ot * 16 + Mb + v;
            ob[(size_t)o * HW] = acc[ot][v] + blin[o];
        }
    }
}

// ---------------- Launcher ----------------------------------------------
extern "C" void kernel_launch(void* const* d_in, const int* in_sizes, int n_in,
                              void* d_out, int out_size, void* d_ws, size_t ws_size,
                              hipStream_t stream) {
    const float* x    = (const float*)d_in[0];
    const int*   spix = (const int*)d_in[1];
    const float* Wl   = (const float*)d_in[2];
    const float* bl   = (const float*)d_in[3];
    const float* wsc  = (const float*)d_in[4];
    const float* bsc  = (const float*)d_in[5];
    float* out = (float*)d_out;

    // Workspace layout (floats): sums | cnts | sq | pwd | scale | Apack(f16)
    float* sums = (float*)d_ws;                 // T*NSP*C      = 25088
    float* cnts = sums + T_ * NSP * C_;         // T*NSP        =   392
    float* sq   = cnts + T_ * NSP;              // T*NSP        =   392
    float* pwd  = sq + T_ * NSP;                // T*NSP*NSP    = 76832
    float* scl  = pwd + T_ * NSP * NSP;         // T*HW         = 51200
    _Float16* Ap = (_Float16*)(scl + T_ * HW);  // byte off 615616, 32B aligned

    hipMemsetAsync(sums, 0, (size_t)(T_ * NSP * C_ + T_ * NSP) * sizeof(float), stream);

    k_segsum<<<(T_ * HW + 255) / 256, 256, 0, stream>>>(x, spix, sums, cnts);
    k_down<<<(T_ * NSP + 127) / 128, 128, 0, stream>>>(sums, cnts, sq);
    k_pwd<<<(T_ * NSP * NSP + 255) / 256, 256, 0, stream>>>(sums, sq, pwd);
    k_scale<<<(T_ * HW + 255) / 256, 256, 0, stream>>>(x, wsc, bsc, scl);
    k_pack<<<(4 * NKC * 32 * 16 + 255) / 256, 256, 0, stream>>>(Wl, Ap);

    dim3 grid(W_ / 16, H_, T_);
    k_conv<<<grid, 32, 0, stream>>>(x, spix, pwd, scl, Ap, bl, out);
}
